// ToyGatedMoE_50070728737584
// MI455X (gfx1250) — compile-verified, matrix operand order checked by reading, not run on hardware
//
#include <hip/hip_runtime.h>
#include <stdint.h>

#define HID 1024
#define NEXP 8
#define BM 128
#define BN 128
#define BK 32
#define AS 40   // A-tile stride (bf16): 80B rows -> aligned b128, conflict-free frag reads
#define BPS 20  // B-tile stride (u32):  80B rows -> aligned b128, conflict-free frag reads

typedef __attribute__((ext_vector_type(16))) __bf16 bf16x16;
typedef __attribute__((ext_vector_type(8)))  float  f32x8;
typedef __attribute__((ext_vector_type(4)))  unsigned int u32x4;
typedef __attribute__((ext_vector_type(4)))  int  i32x4;
typedef __attribute__((ext_vector_type(8)))  int  i32x8;

union Frag16 { uint32_t u[8]; uint4 q[2]; bf16x16 v; };

#if defined(__has_builtin)
#if __has_builtin(__builtin_amdgcn_tensor_load_to_lds) && __has_builtin(__builtin_amdgcn_s_wait_tensorcnt)
#define USE_TDM 1
#endif
#endif
#ifndef USE_TDM
#define USE_TDM 0
#endif

__device__ __forceinline__ uint32_t f2bf_pair(float a, float b) {
  uint32_t ua = __float_as_uint(a), ub = __float_as_uint(b);
  ua = (ua + 0x7FFFu + ((ua >> 16) & 1u)) >> 16;
  ub = (ub + 0x7FFFu + ((ub >> 16) & 1u)) >> 16;
  return ua | (ub << 16);
}
__device__ __forceinline__ uint16_t f2bf1(float a) {
  uint32_t ua = __float_as_uint(a);
  return (uint16_t)((ua + 0x7FFFu + ((ua >> 16) & 1u)) >> 16);
}

#if USE_TDM
// TDM 2D tile: 128 rows (n) x 16 u32 (k-pairs), global row stride 512 u32.
// LDS padding: +4 dwords after every 16 dwords -> 80B LDS row stride (= BPS).
__device__ __forceinline__ void tdm_load_b_tile(uint32_t lds_off, const uint32_t* gsrc) {
  uint64_t ga = (uint64_t)(uintptr_t)gsrc;
  u32x4 g0 = {1u, lds_off, (uint32_t)ga,
              ((uint32_t)(ga >> 32) & 0x01FFFFFFu) | 0x80000000u};   // type=2
  i32x8 g1 = {0x06D20000,  // data_size=4B | pad_enable | pad_interval=16dw | pad_amount=4dw
              0x02000000,  // tensor_dim0[15:0]=512 in [31:16]
              0x00800000,  // tensor_dim1[15:0]=128 in [31:16]
              0x00100000,  // tile_dim0=16 in [31:16]
              128,         // tile_dim1=128, tile_dim2=0
              512,         // tensor_dim0_stride lo
              0, 0};
  i32x4 gz = {0, 0, 0, 0};
#if __clang_major__ >= 23
  i32x8 gz8 = {0, 0, 0, 0, 0, 0, 0, 0};
  __builtin_amdgcn_tensor_load_to_lds(g0, g1, gz, gz, gz8, 0);
#else
  __builtin_amdgcn_tensor_load_to_lds(g0, g1, gz, gz, 0);
#endif
}
#endif

// ---------------- K0: zero output + counters ----------------
__global__ void moe_zero_kernel(float* __restrict__ out, size_t n, int* __restrict__ counts) {
  size_t i = (size_t)blockIdx.x * blockDim.x + threadIdx.x;
  if (i < NEXP) counts[i] = 0;
  size_t stride = (size_t)gridDim.x * blockDim.x;
  for (size_t j = i; j < n; j += stride) out[j] = 0.0f;
}

// ---------------- K0b: fp32 -> bf16 row-major (tokens) ----------------
__global__ __launch_bounds__(256)
void conv_bf16_kernel(const float* __restrict__ src, uint16_t* __restrict__ dst, size_t n8) {
  size_t i = (size_t)blockIdx.x * blockDim.x + threadIdx.x;
  if (i >= n8) return;
  const float4* s = (const float4*)(src + i * 8);
  float4 a = s[0], b = s[1];
  uint4 o;
  o.x = f2bf_pair(a.x, a.y); o.y = f2bf_pair(a.z, a.w);
  o.z = f2bf_pair(b.x, b.y); o.w = f2bf_pair(b.z, b.w);
  *(uint4*)(dst + i * 8) = o;
}

// ---------------- K0c: W fp32 -> bf16 pair-packed, n-major [E][H][H/2] u32 ----------------
// LDS-tiled 64x64 transpose so both global reads and writes stay coalesced.
#define TKP 64
#define TN  64
__global__ __launch_bounds__(256)
void conv_wpackT_kernel(const float* __restrict__ W, uint32_t* __restrict__ Wp) {
  __shared__ uint32_t tile[TKP][TN + 1];
  const int e   = blockIdx.z;
  const int kp0 = blockIdx.y * TKP;
  const int n0  = blockIdx.x * TN;
  const float* We = W + (size_t)e * HID * HID;
  uint32_t* Wpe = Wp + (size_t)e * HID * (HID / 2);
  const int tid = threadIdx.x;
  // load+pack: thread -> one k-pair row, 16 n values
  {
    int kp_l = tid >> 2, ns = (tid & 3) << 4;
    const float* lo = We + (size_t)(2 * (kp0 + kp_l)) * HID + n0 + ns;
    const float* hi = lo + HID;
#pragma unroll
    for (int q = 0; q < 4; ++q) {
      float4 l = *(const float4*)(lo + 4 * q);
      float4 h = *(const float4*)(hi + 4 * q);
      tile[kp_l][ns + 4 * q + 0] = f2bf_pair(l.x, h.x);
      tile[kp_l][ns + 4 * q + 1] = f2bf_pair(l.y, h.y);
      tile[kp_l][ns + 4 * q + 2] = f2bf_pair(l.z, h.z);
      tile[kp_l][ns + 4 * q + 3] = f2bf_pair(l.w, h.w);
    }
  }
  __syncthreads();
  // store transposed: thread -> one n row, 16 consecutive k-pairs
  {
    int n_l = tid >> 2, ks = (tid & 3) << 4;
    uint32_t vals[16];
#pragma unroll
    for (int i = 0; i < 16; ++i) vals[i] = tile[ks + i][n_l];
    uint32_t* dst = Wpe + (size_t)(n0 + n_l) * (HID / 2) + kp0 + ks;
#pragma unroll
    for (int q = 0; q < 4; ++q) {
      uint4 o; o.x = vals[4 * q]; o.y = vals[4 * q + 1];
      o.z = vals[4 * q + 2]; o.w = vals[4 * q + 3];
      *(uint4*)(dst + 4 * q) = o;
    }
  }
}

// ---------------- K1: gating (one wave32 per token) ----------------
__global__ __launch_bounds__(256)
void moe_gate_kernel(const float* __restrict__ X, const float* __restrict__ Gw, int T,
                     int* __restrict__ counts, int* __restrict__ routeIdx,
                     float* __restrict__ routeW) {
  int tok  = (int)((blockIdx.x * (size_t)blockDim.x + threadIdx.x) >> 5);
  int lane = threadIdx.x & 31;
  if (tok >= T) return;
  const float* x = X + (size_t)tok * HID;
  float s[NEXP];
#pragma unroll
  for (int e = 0; e < NEXP; ++e) s[e] = 0.0f;
  for (int h = lane; h < HID; h += 32) {
    float xv = x[h];
#pragma unroll
    for (int e = 0; e < NEXP; ++e) s[e] += xv * Gw[e * HID + h];
  }
#pragma unroll
  for (int e = 0; e < NEXP; ++e) {
#pragma unroll
    for (int off = 16; off > 0; off >>= 1) s[e] += __shfl_xor(s[e], off, 32);
  }
  if (lane == 0) {
    float mx = s[0];
#pragma unroll
    for (int e = 1; e < NEXP; ++e) mx = fmaxf(mx, s[e]);
    float ex[NEXP], sum = 0.0f;
#pragma unroll
    for (int e = 0; e < NEXP; ++e) { ex[e] = expf(s[e] - mx); sum += ex[e]; }
    int i0 = 0;
#pragma unroll
    for (int e = 1; e < NEXP; ++e) if (ex[e] > ex[i0]) i0 = e;
    int i1 = (i0 == 0) ? 1 : 0;
#pragma unroll
    for (int e = 0; e < NEXP; ++e) if (e != i0 && ex[e] > ex[i1]) i1 = e;
    float inv = 1.0f / sum;
    atomicAdd(&counts[i0], 1);
    atomicAdd(&counts[i1], 1);
    routeIdx[2 * tok + 0] = i0;  routeIdx[2 * tok + 1] = i1;
    routeW[2 * tok + 0] = ex[i0] * inv;  routeW[2 * tok + 1] = ex[i1] * inv;
  }
}

// ---------------- K2: capacity plan ----------------
__global__ void moe_plan_kernel(const int* __restrict__ counts, int cap,
                                int* __restrict__ keep, int* __restrict__ baseA,
                                int* __restrict__ lenA, int* __restrict__ cursor) {
  if (threadIdx.x == 0 && blockIdx.x == 0) {
    int run = 0;
    for (int e = 0; e < NEXP; ++e) {
      int c = counts[e];
      int k = (c <= cap) ? 1 : 0;
      int l = k ? c : 0;
      keep[e] = k; lenA[e] = l; baseA[e] = run; cursor[e] = run;
      run += l;
    }
  }
}

// ---------------- K3: build compacted per-expert row lists ----------------
__global__ __launch_bounds__(256)
void moe_scatter_kernel(int T, const int* __restrict__ routeIdx, const float* __restrict__ routeW,
                        const int* __restrict__ keep, int* __restrict__ cursor,
                        int* __restrict__ rowTok, float* __restrict__ rowW) {
  int t = blockIdx.x * blockDim.x + threadIdx.x;
  if (t >= T) return;
#pragma unroll
  for (int s = 0; s < 2; ++s) {
    int e = routeIdx[2 * t + s];
    if (keep[e]) {
      int p = atomicAdd(&cursor[e], 1);
      rowTok[p] = t;
      rowW[p] = routeW[2 * t + s];
    }
  }
}

// ---------------- K4/K5: grouped GEMM, TDM-fed B tiles, double-buffered ----------------
template <int MODE>
__global__ __launch_bounds__(256)
void moe_ffn_gemm(const uint16_t* __restrict__ Abf,   // MODE0: xbf [T][H]; MODE1: hbuf [rows][H]
                  const uint32_t* __restrict__ Wp,    // packed n-major [E][H][H/2]
                  const float* __restrict__ bias,
                  const int* __restrict__ lenA, const int* __restrict__ baseA,
                  const int* __restrict__ rowTok, const float* __restrict__ rowW,
                  uint16_t* __restrict__ Hout, float* __restrict__ Out) {
  const int e = blockIdx.z;
  const int len = lenA[e];
  const int rt = blockIdx.y * BM;
  if (rt >= len) return;                 // uniform per block
  const int base = baseA[e];
  const int colBase = blockIdx.x * BN;
  // tile origin in packed W: rows n = colBase.., columns kp
  const uint32_t* WpeCol = Wp + (size_t)e * HID * (HID / 2) + (size_t)colBase * (HID / 2);

  __shared__ __align__(16) uint16_t Atile[2][BM * AS];
  __shared__ __align__(16) uint32_t Btile[2][BM * BPS];

  const int tid = threadIdx.x;
  const int lane = tid & 31;
  const int wave = tid >> 5;
  const int wr = wave >> 1, wc = wave & 1;
  const int hf = lane >> 4, lm = lane & 15;

  // A staging: 2 threads/row, each copies 16 bf16 (32B) raw
  const int ar = tid >> 1;
  const int aseg = (tid & 1) << 4;
  const bool avalid = (rt + ar) < len;
  const uint16_t* asrc = Abf;
  if (avalid) {
    int p = base + rt + ar;
    size_t r = (MODE == 0) ? (size_t)rowTok[p] : (size_t)p;
    asrc = Abf + r * HID + aseg;
  }

  f32x8 acc[2][4] = {};
  uint4 aq0 = {0, 0, 0, 0}, aq1 = {0, 0, 0, 0};

  // ---- prologue: stage tile 0 ----
  if (avalid) {
    const uint4* s4 = (const uint4*)asrc;
    aq0 = s4[0]; aq1 = s4[1];
  }
#if USE_TDM
  if (wave == 0)
    tdm_load_b_tile((uint32_t)(uintptr_t)&Btile[0][0], WpeCol);
#else
  const int bnr = tid >> 1, bseg = (tid & 1) << 3;   // 2 threads per n-row, 8 u32 each
  uint4 bq0, bq1;
  { const uint4* s4 = (const uint4*)(WpeCol + (size_t)bnr * (HID / 2) + bseg);
    bq0 = s4[0]; bq1 = s4[1]; }
#endif

  const int NITER = HID / BK;            // 32
#pragma unroll 1
  for (int i = 0; i < NITER; ++i) {
    const int cur = i & 1;
    { uint4* adst = (uint4*)&Atile[cur][ar * AS + aseg];
      adst[0] = aq0; adst[1] = aq1; }
#if !USE_TDM
    { uint4* bdst = (uint4*)&Btile[cur][bnr * BPS + bseg];
      bdst[0] = bq0; bdst[1] = bq1; }
#endif
    if (i + 1 < NITER) {
      if (avalid) {
        const uint4* s4 = (const uint4*)(asrc + (size_t)(i + 1) * BK);
        aq0 = s4[0]; aq1 = s4[1];
      }
#if USE_TDM
      if (wave == 0)
        tdm_load_b_tile((uint32_t)(uintptr_t)&Btile[cur ^ 1][0],
                        WpeCol + (size_t)(i + 1) * (BK / 2));
#else
      const uint4* s4 = (const uint4*)(WpeCol + (size_t)bnr * (HID / 2) +
                                       (size_t)(i + 1) * (BK / 2) + bseg);
      bq0 = s4[0]; bq1 = s4[1];
#endif
    }
#if USE_TDM
    if (wave == 0) {                     // in-order TDM: <=1 outstanding => tile i landed
      if (i + 1 < NITER) __builtin_amdgcn_s_wait_tensorcnt(1);
      else               __builtin_amdgcn_s_wait_tensorcnt(0);
    }
#endif
    __syncthreads();

    // ---- fragments: 2x ds_load_b128 each, straight into WMMA operands ----
    Frag16 fa[2], fb[4];
#pragma unroll
    for (int r = 0; r < 2; ++r) {
      const uint16_t* arp = &Atile[cur][(wr * 32 + r * 16 + lm) * AS + hf * 8];
      fa[r].q[0] = *(const uint4*)arp;
      fa[r].q[1] = *(const uint4*)(arp + 16);
    }
#pragma unroll
    for (int c = 0; c < 4; ++c) {
      const uint32_t* brp = &Btile[cur][(wc * 64 + c * 16 + lm) * BPS + hf * 4];
      fb[c].q[0] = *(const uint4*)brp;
      fb[c].q[1] = *(const uint4*)(brp + 8);
    }
#pragma unroll
    for (int r = 0; r < 2; ++r)
#pragma unroll
      for (int c = 0; c < 4; ++c)
        acc[r][c] = __builtin_amdgcn_wmma_f32_16x16x32_bf16(
            false, fa[r].v, false, fb[c].v, (short)0, acc[r][c], false, false);
    __syncthreads();
  }

  // ---- epilogue: C/D layout lane=N%16, VGPR j -> M = half*8 + j ----
#pragma unroll
  for (int r = 0; r < 2; ++r) {
#pragma unroll
    for (int j = 0; j < 8; ++j) {
      int lrow = rt + wr * 32 + r * 16 + hf * 8 + j;
      if (lrow < len) {
        int p = base + lrow;
        if (MODE == 0) {
          size_t ro = (size_t)p * HID;
#pragma unroll
          for (int c = 0; c < 4; ++c) {
            int n = colBase + wc * 64 + c * 16 + lm;
            float v = acc[r][c][j] + bias[e * HID + n];
            v = v > 0.0f ? v : 0.0f;
            Hout[ro + n] = f2bf1(v);
          }
        } else {
          int t = rowTok[p];
          float w = rowW[p];
          size_t ro = (size_t)t * HID;
#pragma unroll
          for (int c = 0; c < 4; ++c) {
            int n = colBase + wc * 64 + c * 16 + lm;
            float v = acc[r][c][j] + bias[e * HID + n];
            atomicAdd(&Out[ro + n], w * v);
          }
        }
      }
    }
  }
}

// ---------------- host launch ----------------
extern "C" void kernel_launch(void* const* d_in, const int* in_sizes, int n_in,
                              void* d_out, int out_size, void* d_ws, size_t ws_size,
                              hipStream_t stream) {
  const float* tokens = (const float*)d_in[0];
  const float* gate_w = (const float*)d_in[1];
  const float* W1 = (const float*)d_in[2];
  const float* b1 = (const float*)d_in[3];
  const float* W2 = (const float*)d_in[4];
  const float* b2 = (const float*)d_in[5];
  float* out = (float*)d_out;

  const int T = in_sizes[0] / HID;             // 8192
  const int cap = (T * 5) / (4 * NEXP);        // int(1.25*T/E) = 1280

  char* w = (char*)d_ws;
  size_t o = 0;
  int* counts = (int*)(w + 0);
  int* keep   = (int*)(w + 32);
  int* baseA  = (int*)(w + 64);
  int* lenA   = (int*)(w + 96);
  int* cursor = (int*)(w + 128);
  o = 256;
  int*      routeIdx = (int*)(w + o);      o += (size_t)8 * T;
  float*    routeW   = (float*)(w + o);    o += (size_t)8 * T;
  int*      rowTok   = (int*)(w + o);      o += (size_t)8 * T;
  float*    rowW     = (float*)(w + o);    o += (size_t)8 * T;
  uint16_t* xbf      = (uint16_t*)(w + o); o += (size_t)2 * T * HID;          // 16MB
  uint16_t* hbuf     = (uint16_t*)(w + o); o += (size_t)4 * T * HID;          // 32MB
  uint32_t* wp1      = (uint32_t*)(w + o); o += (size_t)NEXP * HID * HID * 2; // 16MB
  uint32_t* wp2      = (uint32_t*)(w + o);

  moe_zero_kernel<<<8192, 256, 0, stream>>>(out, (size_t)T * HID, counts);
  {
    size_t n8 = (size_t)T * HID / 8;
    conv_bf16_kernel<<<(unsigned)((n8 + 255) / 256), 256, 0, stream>>>(tokens, xbf, n8);
    dim3 wgrid(HID / TN, (HID / 2) / TKP, NEXP);
    conv_wpackT_kernel<<<wgrid, 256, 0, stream>>>(W1, wp1);
    conv_wpackT_kernel<<<wgrid, 256, 0, stream>>>(W2, wp2);
  }
  moe_gate_kernel<<<(T * 32 + 255) / 256, 256, 0, stream>>>(tokens, gate_w, T, counts,
                                                            routeIdx, routeW);
  moe_plan_kernel<<<1, 32, 0, stream>>>(counts, cap, keep, baseA, lenA, cursor);
  moe_scatter_kernel<<<(T + 255) / 256, 256, 0, stream>>>(T, routeIdx, routeW, keep, cursor,
                                                          rowTok, rowW);
  const int yTiles = (cap + BM - 1) / BM;
  dim3 grid(HID / BN, yTiles, NEXP);
  moe_ffn_gemm<0><<<grid, 256, 0, stream>>>(xbf, wp1, b1, lenA, baseA,
                                            rowTok, rowW, hbuf, nullptr);
  moe_ffn_gemm<1><<<grid, 256, 0, stream>>>(hbuf, wp2, b2, lenA, baseA,
                                            rowTok, rowW, nullptr, out);
}
